// LVQ_57784490000971
// MI455X (gfx1250) — matrix operand order repaired and checked by reading, not compiled, not run on hardware
//
#include <hip/hip_runtime.h>

// ---------------------------------------------------------------------------
// LVQ nearest-codebook classification for MI455X (gfx1250).
//   N=65536 points, D=512, K=1024 codes, classes = codebook_classes[argmin].
// Strategy:
//   * argmin_k (|x|^2 - 2 x.c + |c|^2) == argmin_k (|c|^2 - 2 x.c)  -> drop |x|^2
//   * cross term via v_wmma_f32_16x16x32_bf16 with hi/lo bf16 split
//     (x ~= hi+lo, c ~= hi+lo; keep hi*hi + hi*lo + lo*hi) for ~fp32 fidelity
//   * codebook converted once to bf16 hi/lo (2 MB -> resident in 192 MB L2)
//   * x tile staged in LDS; preferred path: Tensor Data Mover (TDM)
//     tensor_load_to_lds of pre-split bf16 x tiles + s_wait_tensorcnt
// ---------------------------------------------------------------------------

typedef __bf16 bf16_t;
typedef __attribute__((ext_vector_type(4)))  __bf16 v4bf;
typedef __attribute__((ext_vector_type(8)))  __bf16 v8bf;
typedef __attribute__((ext_vector_type(16))) __bf16 v16bf;
typedef __attribute__((ext_vector_type(8)))  float  v8f;
typedef __attribute__((ext_vector_type(4)))  unsigned int v4u;
typedef __attribute__((ext_vector_type(8)))  int v8i;
typedef __attribute__((ext_vector_type(4)))  int v4i;

#define LVQ_N 65536
#define LVQ_D 512
#define LVQ_K 1024

// ---------------------------------------------------------------------------
// Kernel 1: codebook f32 -> bf16 hi/lo split + |c_k|^2 (exact f32).
// ---------------------------------------------------------------------------
__global__ __launch_bounds__(128)
void lvq_prep(const float* __restrict__ cb,
              bf16_t* __restrict__ cb_hi,
              bf16_t* __restrict__ cb_lo,
              float* __restrict__ csq) {
  const int k = blockIdx.x;
  const int t = threadIdx.x;
  const float* row = cb + (size_t)k * LVQ_D;
  float s = 0.0f;
  #pragma unroll
  for (int i = t; i < LVQ_D; i += 128) {
    float f = row[i];
    s += f * f;
    bf16_t h = (bf16_t)f;
    float hf = (float)h;
    bf16_t l = (bf16_t)(f - hf);
    cb_hi[(size_t)k * LVQ_D + i] = h;
    cb_lo[(size_t)k * LVQ_D + i] = l;
  }
  #pragma unroll
  for (int m = 16; m >= 1; m >>= 1) s += __shfl_xor(s, m, 32);
  __shared__ float acc4[4];
  if ((t & 31) == 0) acc4[t >> 5] = s;
  __syncthreads();
  if (t == 0) csq[k] = acc4[0] + acc4[1] + acc4[2] + acc4[3];
}

// ---------------------------------------------------------------------------
// Kernel 2 (TDM path only): stream-convert x (f32) -> x_hi/x_lo (bf16).
// 33.5M elements, 4 per thread, fully coalesced.
// ---------------------------------------------------------------------------
__global__ __launch_bounds__(256)
void lvq_xconv(const float* __restrict__ x,
               bf16_t* __restrict__ x_hi,
               bf16_t* __restrict__ x_lo) {
  const size_t i4 = ((size_t)blockIdx.x * 256 + threadIdx.x) * 4;
  float4 f = *(const float4*)(x + i4);
  v4bf h, l;
  h.x = (bf16_t)f.x; l.x = (bf16_t)(f.x - (float)h.x);
  h.y = (bf16_t)f.y; l.y = (bf16_t)(f.y - (float)h.y);
  h.z = (bf16_t)f.z; l.z = (bf16_t)(f.z - (float)h.z);
  h.w = (bf16_t)f.w; l.w = (bf16_t)(f.w - (float)h.w);
  *(v4bf*)(x_hi + i4) = h;
  *(v4bf*)(x_lo + i4) = l;
}

// ---------------------------------------------------------------------------
// TDM: issue tensor_load_to_lds of a 64x512 bf16 tile (row-major, contiguous)
// into LDS at lds_off. Descriptor per CDNA5 ISA.
// This toolchain's builtin is the 6-arg form:
//   (uint32x4 g0, int32x8 g1, int32x4 g2, int32x4 g3, int32x8 g4, i32 cpol)
// ---------------------------------------------------------------------------
__device__ __forceinline__ void tdm_load_tile_64x512_bf16(const void* gptr,
                                                          unsigned lds_off) {
  const unsigned long long ga = (unsigned long long)(size_t)gptr;
  v4u g0;
  g0[0] = 1u;                                  // count=1, user descriptor
  g0[1] = lds_off;                             // lds_addr (bytes)
  g0[2] = (unsigned)ga;                        // global_addr[31:0]
  g0[3] = (unsigned)(ga >> 32) | (2u << 30);   // global_addr[56:32] | type=2
  v8i g1;
  g1[0] = (int)(1u << 16);                     // wg_mask=0, data_size=1 (2B)
  g1[1] = (int)(512u << 16);                   // tensor_dim0 = 512
  g1[2] = (int)(64u << 16);                    // tensor_dim1 = 64
  g1[3] = (int)(512u << 16);                   // tile_dim0 = 512 (dim1 hi16=0)
  g1[4] = 64;                                  // tile_dim1 = 64, tile_dim2 = 0
  g1[5] = 512;                                 // tensor_dim0_stride = 512
  g1[6] = 0;
  g1[7] = 0;
  v4i z4 = {0, 0, 0, 0};                       // groups 2/3 unused (2D tensor)
  v8i z8 = {0, 0, 0, 0, 0, 0, 0, 0};           // unused extra group
  __builtin_amdgcn_tensor_load_to_lds(g0, g1, z4, z4, z8, 0);
}

// ---------------------------------------------------------------------------
// Fragment loaders (layouts per CDNA5 ISA 7.12.2, wave32).
// A (16x32 bf16): lane<16 -> K 0..7 & 16..23 ; lane>=16 -> K 8..15 & 24..31.
// B (32x16 bf16): lane<16 -> K 0..7 & 16..23 ; lane>=16 -> K 8..15 & 24..31.
// ---------------------------------------------------------------------------
__device__ __forceinline__ v16bf load_a_frag(const bf16_t* p) {
  v8bf a = *(const v8bf*)(p);
  v8bf b = *(const v8bf*)(p + 16);
  return __builtin_shufflevector(a, b, 0, 1, 2, 3, 4, 5, 6, 7,
                                       8, 9, 10, 11, 12, 13, 14, 15);
}
__device__ __forceinline__ v16bf load_b_frag(const bf16_t* p) {
  v8bf a = *(const v8bf*)(p);
  v8bf b = *(const v8bf*)(p + 16);
  return __builtin_shufflevector(a, b, 0, 1, 2, 3, 4, 5, 6, 7,
                                       8, 9, 10, 11, 12, 13, 14, 15);
}

// ---------------------------------------------------------------------------
// Shared compute: 8 wave32s = 4 row-strips x 2 column-halves over the
// 64-row LDS tile; WMMA K-loop, running argmin, lane butterfly, LDS combine.
// ---------------------------------------------------------------------------
__device__ __forceinline__ void lvq_compute(const bf16_t* __restrict__ xhi,
                                            const bf16_t* __restrict__ xlo,
                                            float* redv, int* redi,
                                            const bf16_t* __restrict__ cb_hi,
                                            const bf16_t* __restrict__ cb_lo,
                                            const float* __restrict__ csq,
                                            const int* __restrict__ classes,
                                            int* __restrict__ out,
                                            size_t rowbase) {
  const int tid   = threadIdx.x;
  const int lane  = tid & 31;
  const int wave  = tid >> 5;
  const int strip = wave & 3;   // 16-row strip
  const int half  = wave >> 2;  // column half of K
  const int lmod  = lane & 15;
  const int lhalf = lane >> 4;

  float runval[8];
  int   runidx[8];
  #pragma unroll
  for (int v = 0; v < 8; ++v) { runval[v] = 3.4e38f; runidx[v] = 0; }

  const bf16_t* arow_h = xhi + (strip * 16 + lmod) * LVQ_D + lhalf * 8;
  const bf16_t* arow_l = xlo + (strip * 16 + lmod) * LVQ_D + lhalf * 8;

  const int c0 = half * 512;                 // 32 chunks of 16 columns
  for (int chunk = 0; chunk < 32; ++chunk) {
    const int col = c0 + chunk * 16 + lmod;
    const bf16_t* bh = cb_hi + (size_t)col * LVQ_D + lhalf * 8;
    const bf16_t* bl = cb_lo + (size_t)col * LVQ_D + lhalf * 8;
    __builtin_prefetch(bh + 16 * LVQ_D, 0, 1);
    __builtin_prefetch(bl + 16 * LVQ_D, 0, 1);
    const float cs = csq[col];

    v8f acc = {};
    #pragma unroll 4
    for (int dc = 0; dc < 16; ++dc) {
      v16bf ah  = load_a_frag(arow_h + dc * 32);
      v16bf al  = load_a_frag(arow_l + dc * 32);
      v16bf bhf = load_b_frag(bh + dc * 32);
      v16bf blf = load_b_frag(bl + dc * 32);
      acc = __builtin_amdgcn_wmma_f32_16x16x32_bf16(false, ah, false, bhf,
                                                    (short)0, acc, false, false);
      acc = __builtin_amdgcn_wmma_f32_16x16x32_bf16(false, ah, false, blf,
                                                    (short)0, acc, false, false);
      acc = __builtin_amdgcn_wmma_f32_16x16x32_bf16(false, al, false, bhf,
                                                    (short)0, acc, false, false);
    }

    #pragma unroll
    for (int v = 0; v < 8; ++v) {
      float d2 = cs - 2.0f * acc[v];
      if (d2 < runval[v]) { runval[v] = d2; runidx[v] = col; }
    }
  }

  #pragma unroll
  for (int m = 1; m < 16; m <<= 1) {
    #pragma unroll
    for (int v = 0; v < 8; ++v) {
      float ov = __shfl_xor(runval[v], m, 16);
      int   oi = __shfl_xor(runidx[v], m, 16);
      if (ov < runval[v] || (ov == runval[v] && oi < runidx[v])) {
        runval[v] = ov; runidx[v] = oi;
      }
    }
  }
  if (lmod == 0) {
    #pragma unroll
    for (int v = 0; v < 8; ++v) {
      int r = strip * 16 + lhalf * 8 + v;
      redv[half * 64 + r] = runval[v];
      redi[half * 64 + r] = runidx[v];
    }
  }
  __syncthreads();

  if (tid < 64) {
    float v0 = redv[tid], v1 = redv[64 + tid];
    int   i0 = redi[tid], i1 = redi[64 + tid];
    int best = (v1 < v0 || (v1 == v0 && i1 < i0)) ? i1 : i0;
    out[rowbase + tid] = classes[best];
  }
}

// ---------------------------------------------------------------------------
// Main kernel, TDM path: x tiles already bf16-split in global memory; wave 0
// issues two TENSOR_LOAD_TO_LDS DMAs, waits s_wait_tensorcnt, barrier, compute.
// Dynamic LDS: xhi 64 KB | xlo 64 KB | redv 512 B | redi 512 B = 132096 B.
// ---------------------------------------------------------------------------
__global__ __launch_bounds__(256)
void lvq_main_tdm(const bf16_t* __restrict__ x_hi,
                  const bf16_t* __restrict__ x_lo,
                  const bf16_t* __restrict__ cb_hi,
                  const bf16_t* __restrict__ cb_lo,
                  const float* __restrict__ csq,
                  const int* __restrict__ classes,
                  int* __restrict__ out) {
  extern __shared__ char smem[];
  bf16_t* xhi = (bf16_t*)(smem);
  bf16_t* xlo = (bf16_t*)(smem + 64 * LVQ_D * 2);
  float*  redv = (float*)(smem + 2 * 64 * LVQ_D * 2);
  int*    redi = (int*)(smem + 2 * 64 * LVQ_D * 2 + 512);

  const size_t rowbase = (size_t)blockIdx.x * 64;

  if (threadIdx.x < 32) {   // wave 0 drives the TDM (uniform branch)
    tdm_load_tile_64x512_bf16(x_hi + rowbase * LVQ_D,
                              (unsigned)(size_t)(void*)xhi);
    tdm_load_tile_64x512_bf16(x_lo + rowbase * LVQ_D,
                              (unsigned)(size_t)(void*)xlo);
    __builtin_amdgcn_s_wait_tensorcnt(0);
  }
  __syncthreads();

  lvq_compute(xhi, xlo, redv, redi, cb_hi, cb_lo, csq, classes, out, rowbase);
}

// ---------------------------------------------------------------------------
// Main kernel, fallback path (small workspace): stage + convert x in-kernel.
// ---------------------------------------------------------------------------
__global__ __launch_bounds__(256)
void lvq_main_fb(const float* __restrict__ x,
                 const bf16_t* __restrict__ cb_hi,
                 const bf16_t* __restrict__ cb_lo,
                 const float* __restrict__ csq,
                 const int* __restrict__ classes,
                 int* __restrict__ out) {
  extern __shared__ char smem[];
  bf16_t* xhi = (bf16_t*)(smem);
  bf16_t* xlo = (bf16_t*)(smem + 64 * LVQ_D * 2);
  float*  redv = (float*)(smem + 2 * 64 * LVQ_D * 2);
  int*    redi = (int*)(smem + 2 * 64 * LVQ_D * 2 + 512);

  const int tid = threadIdx.x;
  const size_t rowbase = (size_t)blockIdx.x * 64;

  {
    const int r = tid >> 2;
    const int part = tid & 3;
    const float4* src = (const float4*)(x + (rowbase + r) * LVQ_D + part * 128);
    bf16_t* dh = xhi + r * LVQ_D + part * 128;
    bf16_t* dl = xlo + r * LVQ_D + part * 128;
    #pragma unroll
    for (int i = 0; i < 32; ++i) {
      float4 f = src[i];
      v4bf h, l;
      h.x = (bf16_t)f.x; l.x = (bf16_t)(f.x - (float)h.x);
      h.y = (bf16_t)f.y; l.y = (bf16_t)(f.y - (float)h.y);
      h.z = (bf16_t)f.z; l.z = (bf16_t)(f.z - (float)h.z);
      h.w = (bf16_t)f.w; l.w = (bf16_t)(f.w - (float)h.w);
      *(v4bf*)(dh + 4 * i) = h;
      *(v4bf*)(dl + 4 * i) = l;
    }
  }
  __syncthreads();

  lvq_compute(xhi, xlo, redv, redi, cb_hi, cb_lo, csq, classes, out, rowbase);
}

// ---------------------------------------------------------------------------
// Launcher. Inputs: x (N*D f32), codebook_vectors (K*D f32),
// codebook_classes (K int32). Output: N int32 class ids.
// Workspace: cb_hi 1MB | cb_lo 1MB | csq 4KB | (TDM path) x_hi 64MB | x_lo 64MB.
// Dispatch on ws_size is deterministic (ws_size is fixed per run).
// ---------------------------------------------------------------------------
extern "C" void kernel_launch(void* const* d_in, const int* in_sizes, int n_in,
                              void* d_out, int out_size, void* d_ws, size_t ws_size,
                              hipStream_t stream) {
  const float* x       = (const float*)d_in[0];
  const float* cb      = (const float*)d_in[1];
  const int*   classes = (const int*)d_in[2];
  int* out = (int*)d_out;

  char* ws = (char*)d_ws;
  const size_t cb_bytes = (size_t)LVQ_K * LVQ_D * 2;       // 1 MB each
  const size_t x_bytes  = (size_t)LVQ_N * LVQ_D * 2;       // 64 MB each
  bf16_t* cb_hi = (bf16_t*)(ws);
  bf16_t* cb_lo = (bf16_t*)(ws + cb_bytes);
  float*  csq   = (float*)(ws + 2 * cb_bytes);
  const size_t x_off = 4u << 20;                           // 4 MB aligned base
  bf16_t* x_hi = (bf16_t*)(ws + x_off);
  bf16_t* x_lo = (bf16_t*)(ws + x_off + x_bytes);
  const size_t need_tdm = x_off + 2 * x_bytes;             // ~132 MB

  lvq_prep<<<LVQ_K, 128, 0, stream>>>(cb, cb_hi, cb_lo, csq);

  const size_t smem = (size_t)2 * 64 * LVQ_D * 2 + 512 + 512;  // 132096 B

  if (ws_size >= need_tdm) {
    lvq_xconv<<<(LVQ_N * LVQ_D) / (256 * 4), 256, 0, stream>>>(x, x_hi, x_lo);
    lvq_main_tdm<<<LVQ_N / 64, 256, smem, stream>>>(x_hi, x_lo, cb_hi, cb_lo,
                                                    csq, classes, out);
  } else {
    lvq_main_fb<<<LVQ_N / 64, 256, smem, stream>>>(x, cb_hi, cb_lo,
                                                   csq, classes, out);
  }
}
